// VanillaLSTM_87265145520615
// MI455X (gfx1250) — matrix operand order repaired and checked by reading, not compiled, not run on hardware
//
#include <hip/hip_runtime.h>
#include <cmath>

// ---------------------------------------------------------------------------
// CDNA5 (gfx1250) 2-layer LSTM, wave32, v_wmma_f32_16x16x32_f16.
// f16 operands pre-converted + padded -> b128 fragment loads; streaming gate
// buffers use non-temporal hints so the 192MB L2 keeps weights + h1 resident.
// ---------------------------------------------------------------------------

typedef __attribute__((ext_vector_type(16))) _Float16 v16h;
typedef __attribute__((ext_vector_type(8)))  _Float16 v8h;
typedef __attribute__((ext_vector_type(8)))  float    v8f;

#define TT      512
#define BB      1024
#define IN_DIM  178
#define KPAD    192     // IN_DIM padded to a multiple of 32
#define H1_DIM  64
#define H2_DIM  32

__device__ __forceinline__ v8f wmma_f16(v16h a, v16h b, v8f c) {
  return __builtin_amdgcn_wmma_f32_16x16x32_f16(false, a, false, b, (short)0, c,
                                                false, false);
}

__device__ __forceinline__ v8f zero_v8f() {
  v8f z;
#pragma unroll
  for (int i = 0; i < 8; ++i) z[i] = 0.0f;
  return z;
}

__device__ __forceinline__ float sigmoidf_(float x) {
  return 1.0f / (1.0f + __expf(-x));
}

// ---- WMMA fragment loaders (contiguous k order, same in A and B, wave32) ----------

// A fragment 16x32 f16 from row-major f16 (rows 16-B aligned).
// lane<16: M=lane, K=0..15; lane>=16: M=lane-16, K=16..31.
__device__ __forceinline__ v16h a_frag_f16v(const _Float16* A, int lda, int kb,
                                            int lane) {
  const _Float16* p =
      A + (size_t)(lane & 15) * lda + kb + ((lane & 16) ? 16 : 0);
  v8h lo = *(const v8h*)(p);
  v8h hi = *(const v8h*)(p + 8);
  return __builtin_shufflevector(lo, hi, 0, 1, 2, 3, 4, 5, 6, 7, 8, 9, 10, 11,
                                 12, 13, 14, 15);
}

// Same, but non-temporal (streamed-once A operand, e.g. xh).
__device__ __forceinline__ v16h a_frag_f16v_nt(const _Float16* A, int lda,
                                               int kb, int lane) {
  const _Float16* p =
      A + (size_t)(lane & 15) * lda + kb + ((lane & 16) ? 16 : 0);
  v8h lo = __builtin_nontemporal_load((const v8h*)(p));
  v8h hi = __builtin_nontemporal_load((const v8h*)(p + 8));
  return __builtin_shufflevector(lo, hi, 0, 1, 2, 3, 4, 5, 6, 7, 8, 9, 10, 11,
                                 12, 13, 14, 15);
}

// B fragment 32x16 f16: B[k][n] = W[n_base+n][k], W row-major (NxK) f16.
// lanes 0-15: N=lane, K=kb..kb+15; lanes 16-31: N=lane-16, K=kb+16..kb+31.
__device__ __forceinline__ v16h b_frag_f16v(const _Float16* W, int ldw,
                                            int n_base, int kb, int lane) {
  const _Float16* p =
      W + (size_t)(n_base + (lane & 15)) * ldw + kb + ((lane & 16) ? 16 : 0);
  v8h lo = *(const v8h*)(p);
  v8h hi = *(const v8h*)(p + 8);
  return __builtin_shufflevector(lo, hi, 0, 1, 2, 3, 4, 5, 6, 7, 8, 9, 10, 11,
                                 12, 13, 14, 15);
}

// B fragment from f32 weights (scalar + cvt) — used ONCE per kernel to hoist
// loop-invariant recurrent weights into registers.
__device__ __forceinline__ v16h b_frag_f32(const float* W, int ldw, int n_base,
                                           int kb, int lane) {
  const float* row = W + (size_t)(n_base + (lane & 15)) * ldw;
  const int    kh  = (lane & 16) ? 16 : 0;
  v16h b;
#pragma unroll
  for (int i = 0; i < 8; ++i) {
    int k0       = kb + kh + 2 * i;
    b[2 * i]     = (_Float16)row[k0];
    b[2 * i + 1] = (_Float16)row[k0 + 1];
  }
  return b;
}

// C/D fragment (16x16 f32): VGPR i -> lanes 0-15: M=i, N=lane; lanes16-31: M=i+8.
__device__ __forceinline__ v8f c_frag_load_nt(const float* C, int ldc,
                                              int n_base, int lane) {
  const int n  = n_base + (lane & 15);
  const int mo = (lane & 16) ? 8 : 0;
  v8f c;
#pragma unroll
  for (int i = 0; i < 8; ++i)
    c[i] = __builtin_nontemporal_load(C + (size_t)(mo + i) * ldc + n);
  return c;
}

__device__ __forceinline__ void d_frag_store(float* D, int ldc, int n_base,
                                             int lane, v8f d) {
  const int n  = n_base + (lane & 15);
  const int mo = (lane & 16) ? 8 : 0;
#pragma unroll
  for (int i = 0; i < 8; ++i) D[(size_t)(mo + i) * ldc + n] = d[i];
}

__device__ __forceinline__ void d_frag_store_nt(float* D, int ldc, int n_base,
                                                int lane, v8f d) {
  const int n  = n_base + (lane & 15);
  const int mo = (lane & 16) ? 8 : 0;
#pragma unroll
  for (int i = 0; i < 8; ++i)
    __builtin_nontemporal_store(d[i], D + (size_t)(mo + i) * ldc + n);
}

// ---------------------------------------------------------------------------
// Conversion kernels (run once; streaming, HBM-bound).
// ---------------------------------------------------------------------------
__global__ void __launch_bounds__(192) convert_x_kernel(
    const float* __restrict__ x, _Float16* __restrict__ xh) {
  const size_t m = blockIdx.x;
  const int    k = threadIdx.x;                    // 0..191
  const _Float16 v =
      (k < IN_DIM) ? (_Float16)__builtin_nontemporal_load(x + m * IN_DIM + k)
                   : (_Float16)0.0f;
  __builtin_nontemporal_store(v, xh + m * KPAD + k);
}

__global__ void __launch_bounds__(256) convert_w_kernel(
    const float* __restrict__ w_ih1, const float* __restrict__ w_ih2,
    _Float16* __restrict__ wih1h, _Float16* __restrict__ wih2h) {
  const int tid = threadIdx.x;
  for (int i = tid; i < 256 * KPAD; i += 256) {
    int g    = i / KPAD;
    int k    = i % KPAD;
    wih1h[i] = (k < IN_DIM) ? (_Float16)w_ih1[g * IN_DIM + k] : (_Float16)0.0f;
  }
  for (int i = tid; i < 128 * 64; i += 256) wih2h[i] = (_Float16)w_ih2[i];
}

// ---------------------------------------------------------------------------
// Kernel: xp1 = xh . wih1h^T + b_ih1 + b_hh1   (M=T*B, N=256, K=192 padded)
// One block = 16-row M tile; 8 waves x 2 N-tiles cover N=256.
// ---------------------------------------------------------------------------
__global__ void __launch_bounds__(256) xproj1_kernel(
    const _Float16* __restrict__ xh, const _Float16* __restrict__ wih1h,
    const float* __restrict__ b_ih1, const float* __restrict__ b_hh1,
    float* __restrict__ xp1) {
  const int       lane   = threadIdx.x & 31;
  const int       wid    = threadIdx.x >> 5;
  const size_t    m_base = (size_t)blockIdx.x * 16;
  const _Float16* Arow   = xh + m_base * KPAD;
  const int       nt0    = wid * 2;
  const int       nt1    = wid * 2 + 1;

  v8f acc0 = zero_v8f();
  v8f acc1 = zero_v8f();
#pragma unroll
  for (int kb = 0; kb < KPAD; kb += 32) {
    v16h a = a_frag_f16v_nt(Arow, KPAD, kb, lane);
    acc0   = wmma_f16(a, b_frag_f16v(wih1h, KPAD, nt0 * 16, kb, lane), acc0);
    acc1   = wmma_f16(a, b_frag_f16v(wih1h, KPAD, nt1 * 16, kb, lane), acc1);
  }
  const int   n0    = nt0 * 16 + (lane & 15);
  const int   n1    = nt1 * 16 + (lane & 15);
  const float bias0 = b_ih1[n0] + b_hh1[n0];
  const float bias1 = b_ih1[n1] + b_hh1[n1];
#pragma unroll
  for (int i = 0; i < 8; ++i) { acc0[i] += bias0; acc1[i] += bias1; }
  d_frag_store_nt(xp1 + m_base * 256, 256, nt0 * 16, lane, acc0);
  d_frag_store_nt(xp1 + m_base * 256, 256, nt1 * 16, lane, acc1);
}

// ---------------------------------------------------------------------------
// Kernel: layer-1 recurrence. 64 blocks x 16 batch rows, persistent over t.
// Recurrent weight fragments hoisted into registers before the scan.
// ---------------------------------------------------------------------------
__global__ void __launch_bounds__(256) lstm1_kernel(
    const float* __restrict__ xp1, const float* __restrict__ w_hh1,
    _Float16* __restrict__ h1out) {
  __shared__ __align__(16) _Float16 hbuf[16 * 64];   // h_prev f16
  __shared__ float                  gbuf[16 * 256];  // gate preactivations
  __shared__ float                  cbuf[16 * 64];   // cell state

  const int tid    = threadIdx.x;
  const int lane   = tid & 31;
  const int wid    = tid >> 5;
  const int b_base = blockIdx.x * 16;

  for (int i = tid; i < 16 * 64; i += 256) {
    hbuf[i] = (_Float16)0.0f;
    cbuf[i] = 0.0f;
  }

  const int nt0 = wid * 2;
  const int nt1 = nt0 + 1;

  // Loop-invariant recurrent weight fragments (w_hh1: 256x64 f32).
  const v16h bw00 = b_frag_f32(w_hh1, 64, nt0 * 16, 0, lane);
  const v16h bw01 = b_frag_f32(w_hh1, 64, nt0 * 16, 32, lane);
  const v16h bw10 = b_frag_f32(w_hh1, 64, nt1 * 16, 0, lane);
  const v16h bw11 = b_frag_f32(w_hh1, 64, nt1 * 16, 32, lane);
  __syncthreads();

  for (int t = 0; t < TT; ++t) {
    const float* xpT  = xp1 + ((size_t)t * BB + b_base) * 256;
    v8f          acc0 = c_frag_load_nt(xpT, 256, nt0 * 16, lane);
    v8f          acc1 = c_frag_load_nt(xpT, 256, nt1 * 16, lane);

    v16h a0 = a_frag_f16v(hbuf, 64, 0, lane);
    v16h a1 = a_frag_f16v(hbuf, 64, 32, lane);
    acc0    = wmma_f16(a0, bw00, acc0);
    acc0    = wmma_f16(a1, bw01, acc0);
    acc1    = wmma_f16(a0, bw10, acc1);
    acc1    = wmma_f16(a1, bw11, acc1);

    d_frag_store(gbuf, 256, nt0 * 16, lane, acc0);
    d_frag_store(gbuf, 256, nt1 * 16, lane, acc1);
    __syncthreads();

#pragma unroll
    for (int q = 0; q < 4; ++q) {                    // 16*64 cells, 256 threads
      int   e  = tid + 256 * q;
      int   bb = e >> 6;
      int   j  = e & 63;
      float ig = sigmoidf_(gbuf[bb * 256 + j]);
      float fg = sigmoidf_(gbuf[bb * 256 + 64 + j]);
      float gg = tanhf(gbuf[bb * 256 + 128 + j]);
      float og = sigmoidf_(gbuf[bb * 256 + 192 + j]);
      float c  = fg * cbuf[bb * 64 + j] + ig * gg;
      float h  = og * tanhf(c);
      cbuf[bb * 64 + j] = c;
      hbuf[bb * 64 + j] = (_Float16)h;                        // recurrence
      h1out[((size_t)t * BB + b_base + bb) * 64 + j] =
          (_Float16)fmaxf(h, 0.0f);                           // relu -> L2
    }
    __syncthreads();
  }
}

// ---------------------------------------------------------------------------
// Kernel: xp2 = relu(h1) . wih2h^T + b_ih2 + b_hh2  (M=T*B, N=128, K=64)
// h1 (64 MB) is L2-resident; keep RT hints for it.
// ---------------------------------------------------------------------------
__global__ void __launch_bounds__(256) xproj2_kernel(
    const _Float16* __restrict__ h1, const _Float16* __restrict__ wih2h,
    const float* __restrict__ b_ih2, const float* __restrict__ b_hh2,
    float* __restrict__ xp2) {
  const int       lane   = threadIdx.x & 31;
  const int       wid    = threadIdx.x >> 5;       // 8 waves = 8 N-tiles
  const size_t    m_base = (size_t)blockIdx.x * 16;
  const _Float16* Arow   = h1 + m_base * 64;
  const int       nt     = wid;

  v8f acc = zero_v8f();
#pragma unroll
  for (int kb = 0; kb < 64; kb += 32) {
    v16h a = a_frag_f16v(Arow, 64, kb, lane);
    acc    = wmma_f16(a, b_frag_f16v(wih2h, 64, nt * 16, kb, lane), acc);
  }
  const int   n    = nt * 16 + (lane & 15);
  const float bias = b_ih2[n] + b_hh2[n];
#pragma unroll
  for (int i = 0; i < 8; ++i) acc[i] += bias;
  d_frag_store_nt(xp2 + m_base * 128, 128, nt * 16, lane, acc);
}

// ---------------------------------------------------------------------------
// Kernel: layer-2 recurrence + sigmoid + dense(2) + sigmoid -> out (T, B*2)
// ---------------------------------------------------------------------------
__global__ void __launch_bounds__(256) lstm2_kernel(
    const float* __restrict__ xp2, const float* __restrict__ w_hh2,
    const float* __restrict__ w_dense, const float* __restrict__ b_dense,
    float* __restrict__ out) {
  __shared__ __align__(16) _Float16 hbuf[16 * 32];
  __shared__ float                  gbuf[16 * 128];
  __shared__ float                  cbuf[16 * 32];
  __shared__ float                  sbuf[16 * 32];  // sigmoid(h) for head
  __shared__ float                  wd[2 * 32];
  __shared__ float                  bd[2];

  const int tid    = threadIdx.x;
  const int lane   = tid & 31;
  const int wid    = tid >> 5;
  const int b_base = blockIdx.x * 16;

  for (int i = tid; i < 16 * 32; i += 256) {
    hbuf[i] = (_Float16)0.0f;
    cbuf[i] = 0.0f;
  }
  if (tid < 64) wd[tid] = w_dense[tid];
  if (tid < 2)  bd[tid] = b_dense[tid];

  const int  nt = wid;                               // 8 waves = N=128
  const v16h bw = b_frag_f32(w_hh2, 32, nt * 16, 0, lane);  // hoisted weights
  __syncthreads();

  for (int t = 0; t < TT; ++t) {
    const float* xpT = xp2 + ((size_t)t * BB + b_base) * 128;
    v8f          acc = c_frag_load_nt(xpT, 128, nt * 16, lane);
    acc = wmma_f16(a_frag_f16v(hbuf, 32, 0, lane), bw, acc);
    d_frag_store(gbuf, 128, nt * 16, lane, acc);
    __syncthreads();

#pragma unroll
    for (int q = 0; q < 2; ++q) {                    // 16*32 cells, 256 thr
      int   e  = tid + 256 * q;
      int   bb = e >> 5;
      int   j  = e & 31;
      float ig = sigmoidf_(gbuf[bb * 128 + j]);
      float fg = sigmoidf_(gbuf[bb * 128 + 32 + j]);
      float gg = tanhf(gbuf[bb * 128 + 64 + j]);
      float og = sigmoidf_(gbuf[bb * 128 + 96 + j]);
      float c  = fg * cbuf[bb * 32 + j] + ig * gg;
      float h  = og * tanhf(c);
      cbuf[bb * 32 + j] = c;
      hbuf[bb * 32 + j] = (_Float16)h;
      sbuf[bb * 32 + j] = sigmoidf_(h);
    }
    __syncthreads();

    if (tid < 32) {                                  // 16 b x 2 outputs
      int   bb = tid >> 1;
      int   o  = tid & 1;
      float s  = bd[o];
#pragma unroll
      for (int j = 0; j < 32; ++j) s += sbuf[bb * 32 + j] * wd[o * 32 + j];
      out[(size_t)t * (BB * 2) + (size_t)(b_base + bb) * 2 + o] = sigmoidf_(s);
    }
  }
}

// ---------------------------------------------------------------------------
extern "C" void kernel_launch(void* const* d_in, const int* in_sizes, int n_in,
                              void* d_out, int out_size, void* d_ws,
                              size_t ws_size, hipStream_t stream) {
  const float* x       = (const float*)d_in[0];
  const float* w_ih1   = (const float*)d_in[1];
  const float* w_hh1   = (const float*)d_in[2];
  const float* b_ih1   = (const float*)d_in[3];
  const float* b_hh1   = (const float*)d_in[4];
  const float* w_ih2   = (const float*)d_in[5];
  const float* w_hh2   = (const float*)d_in[6];
  const float* b_ih2   = (const float*)d_in[7];
  const float* b_hh2   = (const float*)d_in[8];
  const float* w_dense = (const float*)d_in[9];
  const float* b_dense = (const float*)d_in[10];
  float*       out     = (float*)d_out;

  const size_t M = (size_t)TT * BB;                  // 524288 rows
  // Workspace layout (all 16B-aligned):
  float*    xp1   = (float*)d_ws;                    // M*256 f32 (512 MB)
  float*    xp2   = xp1 + M * 256;                   // M*128 f32 (256 MB)
  _Float16* h1    = (_Float16*)(xp2 + M * 128);      // M*64  f16 ( 64 MB)
  _Float16* xh    = h1 + M * 64;                     // M*192 f16 (192 MB)
  _Float16* wih1h = xh + M * KPAD;                   // 256*192 f16
  _Float16* wih2h = wih1h + 256 * KPAD;              // 128*64  f16

  convert_x_kernel<<<dim3((unsigned)M), dim3(KPAD), 0, stream>>>(x, xh);
  convert_w_kernel<<<dim3(1), dim3(256), 0, stream>>>(w_ih1, w_ih2, wih1h,
                                                      wih2h);
  dim3 blk(256);
  xproj1_kernel<<<dim3((unsigned)(M / 16)), blk, 0, stream>>>(xh, wih1h, b_ih1,
                                                              b_hh1, xp1);
  lstm1_kernel<<<dim3(BB / 16), blk, 0, stream>>>(xp1, w_hh1, h1);
  xproj2_kernel<<<dim3((unsigned)(M / 16)), blk, 0, stream>>>(h1, wih2h, b_ih2,
                                                              b_hh2, xp2);
  lstm2_kernel<<<dim3(BB / 16), blk, 0, stream>>>(xp2, w_hh2, w_dense, b_dense,
                                                  out);
}
